// Attention_81741817577592
// MI455X (gfx1250) — matrix operand order repaired and checked by reading, not compiled, not run on hardware
//
#include <hip/hip_runtime.h>
#include <hip/hip_bf16.h>

#define EMB      1024
#define HEADS    16
#define SEQ      2048
#define BATCH    4
#define HEAD_DIM 64
#define HALF     32
#define NTOK     (BATCH * SEQ)

typedef __attribute__((ext_vector_type(16))) _Float16 v16h;
typedef __attribute__((ext_vector_type(8)))  _Float16 v8h;
typedef __attribute__((ext_vector_type(8)))  float    v8f;

// ---------------------------------------------------------------------------
// Fragment loaders (wave32); A and B share one contiguous k order, not the ISA table order.
// A 16x32 (f16): lane L holds row L&15; K = (L>>4)*16 .. +15 (contiguous).
//                Two 16B loads.
// B 32x16 (f16): lane L holds col L&15; K = (L>>4)*16 .. +15 (contiguous).
//                Source must be "Bt" layout: for fixed col, K contiguous.
// C/D 16x16 f32: VGPR i: lanes 0-15 -> (row=i, col=lane), lanes 16-31 -> row=i+8.
// ---------------------------------------------------------------------------
__device__ __forceinline__ v16h load_a_frag(const _Float16* p, int ld) {
  const int lane = threadIdx.x & 31;
  const int row  = lane & 15;
  const int koff = (lane >> 4) * 16;
  const _Float16* base = p + (size_t)row * ld;
  v8h lo = *(const v8h*)(base + koff);
  v8h hi = *(const v8h*)(base + koff + 8);
  v16h a;
#pragma unroll
  for (int j = 0; j < 8; ++j) { a[j] = lo[j]; a[8 + j] = hi[j]; }
  return a;
}

__device__ __forceinline__ v16h load_b_frag(const _Float16* bt, int ld) {
  const int lane = threadIdx.x & 31;
  const int col  = lane & 15;
  const int koff = (lane >> 4) * 16;
  const _Float16* base = bt + (size_t)col * ld + koff;
  v8h lo = *(const v8h*)(base);
  v8h hi = *(const v8h*)(base + 8);
  v16h b;
#pragma unroll
  for (int j = 0; j < 8; ++j) { b[j] = lo[j]; b[8 + j] = hi[j]; }
  return b;
}

__device__ __forceinline__ v8f wmma_f16(v16h a, v16h b, v8f c) {
  return __builtin_amdgcn_wmma_f32_16x16x32_f16(false, a, false, b, (short)0, c,
                                                false, false);
}

// ---------------------------------------------------------------------------
// Kernel 1: projection  out = rope(x @ W^T + b)  -> f16 workspace
// grid: (NTOK/128, EMB/64), block 256 (8 waves, 4x2, each 32x32 = 2x2 frags),
// K-tile 64 (two WMMA k-steps per LDS stage, 24 KB LDS).
// mode 0: Q [B,H,S,D] + rope ; mode 1: K [B,H,S,D] + rope ; mode 2: V^T [B,H,D,S]
// ---------------------------------------------------------------------------
__global__ __launch_bounds__(256) void proj_rope_kernel(
    const float* __restrict__ x, const float* __restrict__ W,
    const float* __restrict__ bias, const float* __restrict__ sinT,
    const float* __restrict__ cosT, _Float16* __restrict__ out, int mode) {
  __shared__ __align__(16) _Float16 Atile[128 * 64];  // 16 KB
  __shared__ __align__(16) _Float16 Btile[64 * 64];   // 8 KB

  const int tid  = threadIdx.x;
  const int lane = tid & 31;
  const int w    = tid >> 5;
  const int wm   = w >> 1;   // 0..3
  const int wn   = w & 1;    // 0..1
  const int tok0 = blockIdx.x * 128;
  const int n0   = blockIdx.y * 64;

  v8f acc[2][2];
#pragma unroll
  for (int i = 0; i < 2; ++i)
#pragma unroll
    for (int j = 0; j < 2; ++j) acc[i][j] = (v8f)0.0f;

  for (int k0 = 0; k0 < EMB; k0 += 64) {
    __syncthreads();
    // Stage A: 128 rows x 64 cols, fp32 -> f16 (16 float4 per row)
#pragma unroll
    for (int it = 0; it < 8; ++it) {
      int slot = tid + it * 256;           // 0..2047
      int r = slot >> 4, c4 = slot & 15;
      float4 f = *(const float4*)(x + (size_t)(tok0 + r) * EMB + k0 + c4 * 4);
      _Float16* d = Atile + r * 64 + c4 * 4;
      d[0] = (_Float16)f.x; d[1] = (_Float16)f.y;
      d[2] = (_Float16)f.z; d[3] = (_Float16)f.w;
    }
    // Stage B (Wt): 64 rows (out features) x 64 cols (k)
#pragma unroll
    for (int it = 0; it < 4; ++it) {
      int slot = tid + it * 256;           // 0..1023
      int r = slot >> 4, c4 = slot & 15;
      float4 f = *(const float4*)(W + (size_t)(n0 + r) * EMB + k0 + c4 * 4);
      _Float16* d = Btile + r * 64 + c4 * 4;
      d[0] = (_Float16)f.x; d[1] = (_Float16)f.y;
      d[2] = (_Float16)f.z; d[3] = (_Float16)f.w;
    }
    __syncthreads();

#pragma unroll
    for (int ks = 0; ks < 2; ++ks) {
      v16h aF[2], bF[2];
#pragma unroll
      for (int i = 0; i < 2; ++i)
        aF[i] = load_a_frag(Atile + (wm * 32 + i * 16) * 64 + ks * 32, 64);
#pragma unroll
      for (int j = 0; j < 2; ++j)
        bF[j] = load_b_frag(Btile + (wn * 32 + j * 16) * 64 + ks * 32, 64);
#pragma unroll
      for (int i = 0; i < 2; ++i)
#pragma unroll
        for (int j = 0; j < 2; ++j) acc[i][j] = wmma_f16(aF[i], bF[j], acc[i][j]);
    }
  }

  // Epilogue: bias, RoPE (via adjacent-lane shuffle), store f16
#pragma unroll
  for (int i = 0; i < 2; ++i) {
#pragma unroll
    for (int j = 0; j < 2; ++j) {
#pragma unroll
      for (int e = 0; e < 8; ++e) {
        int row_l = wm * 32 + i * 16 + e + ((lane >> 4) << 3);
        int f     = n0 + wn * 32 + j * 16 + (lane & 15);
        int tok   = tok0 + row_l;
        int b     = tok / SEQ, s = tok % SEQ;
        float v = acc[i][j][e] + bias[f];
        if (mode < 2) {  // RoPE on Q,K: pairs (2i,2i+1) sit in adjacent lanes
          float partner = __shfl_xor(v, 1, 32);
          int dd = f & (HEAD_DIM - 1);
          int ir = dd >> 1;
          float sn = sinT[s * HALF + ir];
          float cs = cosT[s * HALF + ir];
          v = (dd & 1) ? (partner * sn + v * cs) : (v * cs - partner * sn);
        }
        int h = f >> 6, d = f & 63;
        size_t idx;
        if (mode < 2)  // [B,H,S,D]
          idx = (((size_t)b * HEADS + h) * SEQ + s) * HEAD_DIM + d;
        else           // V transposed: [B,H,D,S]
          idx = (((size_t)b * HEADS + h) * HEAD_DIM + d) * SEQ + s;
        out[idx] = (_Float16)v;
      }
    }
  }
}

// ---------------------------------------------------------------------------
// Kernel 2: causal flash attention.
// grid: (SEQ/256, B*H), block 256 = 8 waves; each wave owns 32 queries
// (2 row-fragments) so every K/V fragment is reused twice.
// Qf,Kf: [B,H,S,D] f16 ; Vt: [B,H,D,S] f16 ; out: [B,S,EMB] f32
// ---------------------------------------------------------------------------
__global__ __launch_bounds__(256) void flash_attn_kernel(
    const _Float16* __restrict__ Qf, const _Float16* __restrict__ Kf,
    const _Float16* __restrict__ Vt, float* __restrict__ out) {
  __shared__ __align__(16) _Float16 pTile[8 * 32 * 32];  // 16 KB, 2 KB / wave

  const int lane = threadIdx.x & 31;
  const int w    = threadIdx.x >> 5;
  const int bh   = blockIdx.y;
  const int b    = bh >> 4;   // /HEADS
  const int h    = bh & 15;
  const int q0   = blockIdx.x * 256 + w * 32;

  const _Float16* Qb = Qf + (size_t)bh * SEQ * HEAD_DIM;
  const _Float16* Kb = Kf + (size_t)bh * SEQ * HEAD_DIM;
  const _Float16* Vb = Vt + (size_t)bh * HEAD_DIM * SEQ;
  _Float16* myP = pTile + w * (32 * 32);

  // Resident Q fragments: 2 row-frags x 2 k-steps over d
  v16h aQ[2][2];
#pragma unroll
  for (int r = 0; r < 2; ++r)
#pragma unroll
    for (int ks = 0; ks < 2; ++ks)
      aQ[r][ks] =
          load_a_frag(Qb + (size_t)(q0 + r * 16) * HEAD_DIM + ks * 32, HEAD_DIM);

  v8f o[2][4];
#pragma unroll
  for (int r = 0; r < 2; ++r)
#pragma unroll
    for (int j = 0; j < 4; ++j) o[r][j] = (v8f)0.0f;
  float m[2][8], l[2][8];
#pragma unroll
  for (int r = 0; r < 2; ++r)
#pragma unroll
    for (int e = 0; e < 8; ++e) { m[r][e] = -1e30f; l[r][e] = 0.0f; }

  const float scale = 0.125f;  // 1/sqrt(64)
  const int kend = q0 + 31;

  for (int key0 = 0; key0 <= kend; key0 += 32) {
    // Prefetch next key block (K rows + V columns) -> global_prefetch
    if (key0 + 32 <= kend) {
      __builtin_prefetch(Kb + (size_t)(key0 + 32) * HEAD_DIM + lane * 4, 0, 1);
      __builtin_prefetch(Vb + (size_t)(lane >> 3) * 16 * SEQ + key0 + 32, 0, 1);
    }

    // ---- scores: 32 queries x 32 keys ----
    v8f sc[2][2];  // [row frag][col frag]
#pragma unroll
    for (int nb = 0; nb < 2; ++nb) {
      v16h bK0 = load_b_frag(Kb + (size_t)(key0 + nb * 16) * HEAD_DIM, HEAD_DIM);
      v16h bK1 =
          load_b_frag(Kb + (size_t)(key0 + nb * 16) * HEAD_DIM + 32, HEAD_DIM);
#pragma unroll
      for (int r = 0; r < 2; ++r) {
        v8f c = wmma_f16(aQ[r][0], bK0, (v8f)0.0f);
        sc[r][nb] = wmma_f16(aQ[r][1], bK1, c);
      }
    }

    // ---- scale + causal mask + online softmax (per row fragment) ----
#pragma unroll
    for (int r = 0; r < 2; ++r) {
#pragma unroll
      for (int e = 0; e < 8; ++e) {
        int rl  = r * 16 + e + ((lane >> 4) << 3);  // row within 32-query tile
        int row = q0 + rl;
        float s0 = sc[r][0][e] * scale, s1 = sc[r][1][e] * scale;
        int c0 = key0 + (lane & 15), c1 = c0 + 16;
        if (c0 > row) s0 = -1e30f;
        if (c1 > row) s1 = -1e30f;
        float rm = fmaxf(s0, s1);
#pragma unroll
        for (int off = 1; off < 16; off <<= 1)
          rm = fmaxf(rm, __shfl_xor(rm, off, 32));
        float mn = fmaxf(m[r][e], rm);
        float alpha = __expf(m[r][e] - mn);
        m[r][e] = mn;
        float p0 = __expf(s0 - mn), p1 = __expf(s1 - mn);
        float rs = p0 + p1;
#pragma unroll
        for (int off = 1; off < 16; off <<= 1) rs += __shfl_xor(rs, off, 32);
        l[r][e] = l[r][e] * alpha + rs;
#pragma unroll
        for (int j = 0; j < 4; ++j) o[r][j][e] *= alpha;
        // stash P (C-layout) into LDS row-major 32x32 for A-frag reload
        myP[rl * 32 + (lane & 15)]      = (_Float16)p0;
        myP[rl * 32 + 16 + (lane & 15)] = (_Float16)p1;
      }
    }

    asm volatile("s_wait_dscnt 0x0" ::: "memory");  // DS stores -> DS loads (same wave)

    // ---- O += P(32x32) @ V(32x64): share each V fragment across row frags ----
#pragma unroll
    for (int db = 0; db < 4; ++db) {
      v16h bV = load_b_frag(Vb + (size_t)(db * 16) * SEQ + key0, SEQ);
#pragma unroll
      for (int r = 0; r < 2; ++r) {
        v16h pa = load_a_frag(myP + r * 16 * 32, 32);
        o[r][db] = wmma_f16(pa, bV, o[r][db]);
      }
    }
  }

  // ---- normalize + store fp32 [B,S,EMB] ----
#pragma unroll
  for (int r = 0; r < 2; ++r) {
#pragma unroll
    for (int e = 0; e < 8; ++e) {
      float inv = 1.0f / l[r][e];
      int srow = q0 + r * 16 + e + ((lane >> 4) << 3);
#pragma unroll
      for (int db = 0; db < 4; ++db) {
        int d = db * 16 + (lane & 15);
        out[((size_t)b * SEQ + srow) * EMB + h * HEAD_DIM + d] = o[r][db][e] * inv;
      }
    }
  }
}

// ---------------------------------------------------------------------------
extern "C" void kernel_launch(void* const* d_in, const int* in_sizes, int n_in,
                              void* d_out, int out_size, void* d_ws, size_t ws_size,
                              hipStream_t stream) {
  const float* q    = (const float*)d_in[0];
  const float* k    = (const float*)d_in[1];
  const float* v    = (const float*)d_in[2];
  const float* Wq   = (const float*)d_in[3];
  const float* bq   = (const float*)d_in[4];
  const float* Wk   = (const float*)d_in[5];
  const float* bk   = (const float*)d_in[6];
  const float* Wv   = (const float*)d_in[7];
  const float* bv   = (const float*)d_in[8];
  const float* sinT = (const float*)d_in[9];
  const float* cosT = (const float*)d_in[10];
  // d_in[11] = mask flag (always 1 / causal in this config)

  const size_t NQ = (size_t)BATCH * HEADS * SEQ * HEAD_DIM;  // 8.4M f16 each
  _Float16* Qf = (_Float16*)d_ws;
  _Float16* Kf = Qf + NQ;
  _Float16* Vt = Kf + NQ;

  dim3 blk(256);
  dim3 gproj(NTOK / 128, EMB / 64, 1);
  proj_rope_kernel<<<gproj, blk, 0, stream>>>(q, Wq, bq, sinT, cosT, Qf, 0);
  proj_rope_kernel<<<gproj, blk, 0, stream>>>(k, Wk, bk, sinT, cosT, Kf, 1);
  proj_rope_kernel<<<gproj, blk, 0, stream>>>(v, Wv, bv, sinT, cosT, Vt, 2);

  dim3 gattn(SEQ / 256, BATCH * HEADS, 1);
  flash_attn_kernel<<<gattn, blk, 0, stream>>>(Qf, Kf, Vt, (float*)d_out);
}
